// BigBirdSeq2SeqBlock_69020124446911
// MI455X (gfx1250) — matrix operand order repaired and checked by reading, not compiled, not run on hardware
//
#include <hip/hip_runtime.h>
#include <hip/hip_bf16.h>

// ---------------------------------------------------------------------------
// BigBird seq2seq decoder block for MI455X (gfx1250), wave32 + WMMA f16.
// - All matmuls via v_wmma_f32_16x16x32_f16 (f32 accumulate).
// - All GEMM/attention operands stored f16 in workspace; weights pre-
//   transposed+converted once to f16 [N][K] so LDS staging is a pure copy.
// - Tile staging uses the CDNA5 Tensor Data Mover (tensor_load_to_lds +
//   s_wait_tensorcnt) with hardware LDS row padding; ping-pong buffers in
//   the GEMM overlap DMA with WMMA. Falls back to VALU staging if the
//   builtins are unavailable.
// ---------------------------------------------------------------------------

typedef _Float16 half8  __attribute__((ext_vector_type(8)));
typedef _Float16 half16 __attribute__((ext_vector_type(16)));
typedef float    v8f    __attribute__((ext_vector_type(8)));
typedef unsigned int u32x4 __attribute__((ext_vector_type(4)));
typedef int          i32x4 __attribute__((ext_vector_type(4)));
typedef int          i32x8 __attribute__((ext_vector_type(8)));

#define B_   2
#define ST_  2048
#define SK_  2048
#define D_   512
#define H_   8
#define HD_  64
#define MLP_ 2048

#if __has_builtin(__builtin_amdgcn_tensor_load_to_lds) && \
    __has_builtin(__builtin_amdgcn_s_wait_tensorcnt)
#define USE_TDM 1
#else
#define USE_TDM 0
#endif

struct Frag16 { union { half16 v; half8 h[2]; }; };

__device__ __forceinline__ v8f wmma_f16(half16 a, half16 b, v8f c) {
    return __builtin_amdgcn_wmma_f32_16x16x32_f16(false, a, false, b, (short)0, c,
                                                  false, false);
}

__device__ __forceinline__ float gelu_tanh(float x) {
    const float k0 = 0.7978845608028654f;   // sqrt(2/pi)
    const float k1 = 0.044715f;
    float x3 = x * x * x;
    return 0.5f * x * (1.0f + tanhf(k0 * (x + k1 * x3)));
}

#if USE_TDM
__device__ __forceinline__ unsigned lds_off(const void* p) {
    return (unsigned)(unsigned long long)
           (__attribute__((address_space(3))) const void*)p;
}

// 2D f16 tile load: rows of 64 valid f16 elements (row stride rowStride
// elements) -> LDS with 16B padding every 128B (144B padded row stride,
// matching [..][72] _Float16 LDS arrays).
__device__ __forceinline__ void tdm_load_tile(unsigned ldsByteOff,
                                              const _Float16* g,
                                              int tileRows, int tensorRows,
                                              int rowStride) {
    unsigned long long ga = (unsigned long long)g;
    u32x4 g0;
    g0[0] = 1u;                                    // count = 1 valid descriptor
    g0[1] = ldsByteOff;                            // lds_addr (bytes)
    g0[2] = (unsigned)(ga & 0xffffffffu);          // global_addr lo
    g0[3] = (unsigned)(ga >> 32) | 0x80000000u;    // global_addr hi | type=2
    i32x8 g1;
    // data_size=1 (2B), pad_enable, pad_interval=4 (32 DWORDs = 128B),
    // pad_amount=3 (4 DWORDs = 16B), workgroup_mask=0
    g1[0] = (int)((1u << 16) | (1u << 20) | (4u << 22) | (3u << 25));
    g1[1] = (int)(64u << 16);                      // tensor_dim0 = 64 (lo16)
    g1[2] = (int)(((unsigned)tensorRows & 0xffffu) << 16);  // dim1 lo16
    g1[3] = (int)((((unsigned)tensorRows >> 16) & 0xffffu) | (64u << 16)); // dim1 hi | tile_dim0=64
    g1[4] = (int)((unsigned)tileRows & 0xffffu);   // tile_dim1 | tile_dim2=0
    g1[5] = rowStride;                             // tensor_dim0_stride lo
    g1[6] = 0;
    g1[7] = 0;
    i32x4 z4 = {0, 0, 0, 0};
    i32x8 z8 = {0, 0, 0, 0, 0, 0, 0, 0};
    // 6-arg form (clang-23 / therock headers): g0, g1, g2, g3, extra, cpol
    __builtin_amdgcn_tensor_load_to_lds(g0, g1, z4, z4, z8, 0);
}
#endif

// ---------------------------------------------------------------------------
// Weight transpose+convert: Wt[n*K + k] = (f16) W[k*N + n]
// ---------------------------------------------------------------------------
__global__ __launch_bounds__(256) void trconv_kernel(const float* __restrict__ W,
                                                     _Float16* __restrict__ Wt,
                                                     int K, int N) {
    int idx = blockIdx.x * 256 + threadIdx.x;
    int n = idx / K, k = idx - n * K;
    Wt[idx] = (_Float16)W[(size_t)k * N + n];
}

__global__ __launch_bounds__(256) void conv16_kernel(const float* __restrict__ x,
                                                     _Float16* __restrict__ y) {
    int idx = blockIdx.x * 256 + threadIdx.x;
    y[idx] = (_Float16)x[idx];
}

// ---------------------------------------------------------------------------
// LayerNorm over rows of 512, f32 in -> f16 out. One wave per row.
// ---------------------------------------------------------------------------
__global__ __launch_bounds__(256) void ln512_kernel(const float* __restrict__ x,
                                                    const float* __restrict__ s,
                                                    const float* __restrict__ b,
                                                    _Float16* __restrict__ y) {
    int wave = threadIdx.x >> 5;
    int lane = threadIdx.x & 31;
    int row  = blockIdx.x * 8 + wave;
    const float* xr = x + (size_t)row * D_;
    float vals[16];
    float sum = 0.f, sq = 0.f;
#pragma unroll
    for (int i = 0; i < 16; ++i) {
        float v = xr[i * 32 + lane];
        vals[i] = v;
        sum += v;
        sq  += v * v;
    }
#pragma unroll
    for (int off = 1; off < 32; off <<= 1) {
        sum += __shfl_xor(sum, off, 32);
        sq  += __shfl_xor(sq,  off, 32);
    }
    float mean = sum * (1.0f / D_);
    float var  = sq * (1.0f / D_) - mean * mean;
    float rs   = rsqrtf(var + 1e-6f);
    _Float16* yr = y + (size_t)row * D_;
#pragma unroll
    for (int i = 0; i < 16; ++i) {
        int c = i * 32 + lane;
        yr[c] = (_Float16)((vals[i] - mean) * rs * s[c] + b[c]);
    }
}

// ---------------------------------------------------------------------------
// WMMA GEMM: C[M x N] = epilogue(alpha * A[M x K] @ Wt[N x K]^T).
// A, Wt are f16; Wt pre-transposed so both LDS tiles are pure copies.
// Block tile 64x64, K-step 64, ping-pong LDS buffers fed by the TDM.
// MODE: 0 = alpha*acc   1 = acc + R   2 = gelu(acc + bias)   3 = acc+bias+R
// ---------------------------------------------------------------------------
template <int MODE, typename CT>
__global__ __launch_bounds__(128) void gemm64_kernel(const _Float16* __restrict__ A,
                                                     const _Float16* __restrict__ Wt,
                                                     CT* __restrict__ C,
                                                     const float* __restrict__ bias,
                                                     const float* __restrict__ R,
                                                     int K, int N, float alpha) {
    __shared__ _Float16 Ash[2][64][72];   // 144B padded rows (16B aligned)
    __shared__ _Float16 Bsh[2][64][72];   // Wt tile: rows = n, cols = k

    const int tid  = threadIdx.x;
    const int wave = tid >> 5;
    const int lane = tid & 31;
    const int m_   = lane & 15;
    const int half = lane >> 4;

    const int m0 = blockIdx.x * 64;
    const int n0 = blockIdx.y * 64;

    v8f acc[4];
#pragma unroll
    for (int i = 0; i < 4; ++i) acc[i] = (v8f)0.0f;

    auto compute = [&](int buf) {
#pragma unroll
        for (int c = 0; c < 2; ++c) {
            Frag16 a;
            a.h[0] = *(const half8*)&Ash[buf][wave * 16 + m_][c * 32 + half * 8];
            a.h[1] = *(const half8*)&Ash[buf][wave * 16 + m_][c * 32 + 16 + half * 8];
#pragma unroll
            for (int nc = 0; nc < 4; ++nc) {
                Frag16 bf;
                bf.h[0] = *(const half8*)&Bsh[buf][nc * 16 + m_][c * 32 + half * 8];
                bf.h[1] = *(const half8*)&Bsh[buf][nc * 16 + m_][c * 32 + 16 + half * 8];
                acc[nc] = wmma_f16(a.v, bf.v, acc[nc]);
            }
        }
    };

    const int nk = K >> 6;
#if USE_TDM
    if (wave == 0) {
        tdm_load_tile(lds_off(&Ash[0][0][0]), &A[(size_t)m0 * K], 64, 64, K);
        tdm_load_tile(lds_off(&Bsh[0][0][0]), &Wt[(size_t)n0 * K], 64, 64, K);
    }
    for (int ks = 0; ks < nk; ++ks) {
        int cur = ks & 1;
        if (wave == 0) __builtin_amdgcn_s_wait_tensorcnt(0);
        __syncthreads();
        if (wave == 0 && ks + 1 < nk) {
            int k0 = (ks + 1) << 6;
            tdm_load_tile(lds_off(&Ash[cur ^ 1][0][0]), &A[(size_t)m0 * K + k0], 64, 64, K);
            tdm_load_tile(lds_off(&Bsh[cur ^ 1][0][0]), &Wt[(size_t)n0 * K + k0], 64, 64, K);
        }
        compute(cur);
    }
#else
    for (int ks = 0; ks < nk; ++ks) {
        int k0 = ks << 6;
        __syncthreads();
#pragma unroll
        for (int p = 0; p < 4; ++p) {
            int idx = p * 128 + tid;
            int r = idx >> 3, c8 = idx & 7;
            *(half8*)&Ash[0][r][c8 * 8] = *(const half8*)&A[(size_t)(m0 + r) * K + k0 + c8 * 8];
            *(half8*)&Bsh[0][r][c8 * 8] = *(const half8*)&Wt[(size_t)(n0 + r) * K + k0 + c8 * 8];
        }
        __syncthreads();
        compute(0);
    }
#endif

    // epilogue: lane row = r + 8*half, col = nc*16 + m_
#pragma unroll
    for (int nc = 0; nc < 4; ++nc) {
#pragma unroll
        for (int r = 0; r < 8; ++r) {
            int grow = m0 + wave * 16 + r + 8 * half;
            int gcol = n0 + nc * 16 + m_;
            size_t idx = (size_t)grow * N + gcol;
            float v = acc[nc][r];
            if (MODE == 0) {
                v *= alpha;
            } else if (MODE == 1) {
                v += R[idx];
            } else if (MODE == 2) {
                v = gelu_tanh(v + bias[gcol]);
            } else if (MODE == 3) {
                v += bias[gcol] + R[idx];
            }
            C[idx] = (CT)v;
        }
    }
}

// ---------------------------------------------------------------------------
// Flash attention, elementwise bool mask, q pre-scaled by 1/sqrt(HD).
// Block = (b, h, 64 q-rows), 4 waves x 16 q-rows. q/k/v/o: f16 [B,S,H*HD].
// Q and K tiles staged via TDM; V staged transposed by VALU (overlaps DMA).
// ---------------------------------------------------------------------------
__global__ __launch_bounds__(128) void attn64_kernel(const _Float16* __restrict__ q,
                                                     const _Float16* __restrict__ k,
                                                     const _Float16* __restrict__ v,
                                                     _Float16* __restrict__ o,
                                                     const unsigned char* __restrict__ mask,
                                                     int SKlen) {
    __shared__ _Float16 Qsh[64][72];
    __shared__ _Float16 Ksh[32][72];
    __shared__ _Float16 Vt[64][40];        // V transposed: Vt[d][key-in-chunk]
    __shared__ _Float16 Psh[4][16][40];    // per-wave P tile 16x32

    const int tid  = threadIdx.x;
    const int wave = tid >> 5;
    const int lane = tid & 31;
    const int m_   = lane & 15;
    const int half = lane >> 4;

    const int qb = blockIdx.x;
    const int h  = blockIdx.y;
    const int b  = blockIdx.z;
    const int HHD = H_ * HD_;

    const _Float16* qbase = &q[((size_t)(b * ST_ + qb * 64)) * HHD + h * HD_];
#if USE_TDM
    if (wave == 0) {
        tdm_load_tile(lds_off(&Qsh[0][0]), qbase, 64, 64, HHD);
        __builtin_amdgcn_s_wait_tensorcnt(0);
    }
    __syncthreads();
#else
#pragma unroll
    for (int p = 0; p < 4; ++p) {
        int idx = p * 128 + tid;
        int r = idx >> 3, c8 = idx & 7;
        *(half8*)&Qsh[r][c8 * 8] = *(const half8*)&qbase[(size_t)r * HHD + c8 * 8];
    }
    __syncthreads();
#endif

    Frag16 aq[2];
#pragma unroll
    for (int c = 0; c < 2; ++c) {
        aq[c].h[0] = *(const half8*)&Qsh[wave * 16 + m_][c * 32 + half * 8];
        aq[c].h[1] = *(const half8*)&Qsh[wave * 16 + m_][c * 32 + 16 + half * 8];
    }

    float mrun[8], lrun[8];
    v8f accO[4];
#pragma unroll
    for (int r = 0; r < 8; ++r) { mrun[r] = -1e30f; lrun[r] = 0.0f; }
#pragma unroll
    for (int nc = 0; nc < 4; ++nc) accO[nc] = (v8f)0.0f;

    for (int k0 = 0; k0 < SKlen; k0 += 32) {
        __syncthreads();
        const _Float16* kbase = &k[((size_t)(b * SK_ + k0)) * HHD + h * HD_];
        const _Float16* vbase = &v[((size_t)(b * SK_ + k0)) * HHD + h * HD_];
#if USE_TDM
        if (wave == 0) tdm_load_tile(lds_off(&Ksh[0][0]), kbase, 32, SKlen - k0, HHD);
#else
#pragma unroll
        for (int p = 0; p < 2; ++p) {
            int idx = p * 128 + tid;
            int r = idx >> 3, c8 = idx & 7;
            *(half8*)&Ksh[r][c8 * 8] = *(const half8*)&kbase[(size_t)r * HHD + c8 * 8];
        }
#endif
        // V staged transposed (VALU path; overlaps the TDM on K)
#pragma unroll
        for (int p = 0; p < 2; ++p) {
            int idx = p * 128 + tid;
            int r = idx >> 3, c8 = idx & 7;
            half8 hv = *(const half8*)&vbase[(size_t)r * HHD + c8 * 8];
#pragma unroll
            for (int j = 0; j < 8; ++j) Vt[c8 * 8 + j][r] = hv[j];
        }
#if USE_TDM
        if (wave == 0) __builtin_amdgcn_s_wait_tensorcnt(0);
#endif
        __syncthreads();

        // scores for 32 keys: two 16x16 tiles, 2 WMMAs each over d-chunks
        v8f s[2];
#pragma unroll
        for (int kt = 0; kt < 2; ++kt) {
            s[kt] = (v8f)0.0f;
#pragma unroll
            for (int c = 0; c < 2; ++c) {
                Frag16 bf;
                bf.h[0] = *(const half8*)&Ksh[kt * 16 + m_][c * 32 + half * 8];
                bf.h[1] = *(const half8*)&Ksh[kt * 16 + m_][c * 32 + 16 + half * 8];
                s[kt] = wmma_f16(aq[c].v, bf.v, s[kt]);
            }
        }
        // elementwise mask
#pragma unroll
        for (int kt = 0; kt < 2; ++kt) {
#pragma unroll
            for (int r = 0; r < 8; ++r) {
                int qrow = qb * 64 + wave * 16 + r + 8 * half;
                int kcol = k0 + kt * 16 + m_;
                if (!mask[(size_t)qrow * SKlen + kcol]) s[kt][r] = -1e9f;
            }
        }
        // online softmax (rows live within 16-lane halves)
#pragma unroll
        for (int r = 0; r < 8; ++r) {
            float mv = fmaxf(s[0][r], s[1][r]);
#pragma unroll
            for (int off = 1; off < 16; off <<= 1)
                mv = fmaxf(mv, __shfl_xor(mv, off, 32));
            float mnew = fmaxf(mrun[r], mv);
            float scl  = __expf(mrun[r] - mnew);
            mrun[r] = mnew;
            lrun[r] *= scl;
#pragma unroll
            for (int nc = 0; nc < 4; ++nc) accO[nc][r] *= scl;

            float p0 = __expf(s[0][r] - mnew);
            float p1 = __expf(s[1][r] - mnew);
            Psh[wave][r + 8 * half][m_]      = (_Float16)p0;
            Psh[wave][r + 8 * half][16 + m_] = (_Float16)p1;
            float rs = p0 + p1;
#pragma unroll
            for (int off = 1; off < 16; off <<= 1)
                rs += __shfl_xor(rs, off, 32);
            lrun[r] += rs;
        }
        // P (16x32) @ V (32x64)
        Frag16 ap;
        ap.h[0] = *(const half8*)&Psh[wave][m_][half * 8];
        ap.h[1] = *(const half8*)&Psh[wave][m_][16 + half * 8];
#pragma unroll
        for (int nc = 0; nc < 4; ++nc) {
            Frag16 bv;
            bv.h[0] = *(const half8*)&Vt[nc * 16 + m_][half * 8];
            bv.h[1] = *(const half8*)&Vt[nc * 16 + m_][16 + half * 8];
            accO[nc] = wmma_f16(ap.v, bv.v, accO[nc]);
        }
    }

#pragma unroll
    for (int nc = 0; nc < 4; ++nc) {
#pragma unroll
        for (int r = 0; r < 8; ++r) {
            int qrow = qb * 64 + wave * 16 + r + 8 * half;
            float val = accO[nc][r] / fmaxf(lrun[r], 1e-20f);
            o[((size_t)(b * ST_ + qrow)) * HHD + h * HD_ + nc * 16 + m_] = (_Float16)val;
        }
    }
}

// ---------------------------------------------------------------------------
// Host orchestration
// ---------------------------------------------------------------------------
extern "C" void kernel_launch(void* const* d_in, const int* in_sizes, int n_in,
                              void* d_out, int out_size, void* d_ws, size_t ws_size,
                              hipStream_t stream) {
    const float* targets = (const float*)d_in[0];
    const float* encoded = (const float*)d_in[1];
    const float* ln1s = (const float*)d_in[2];
    const float* ln1b = (const float*)d_in[3];
    const float* ln2s = (const float*)d_in[4];
    const float* ln2b = (const float*)d_in[5];
    const float* ln3s = (const float*)d_in[6];
    const float* ln3b = (const float*)d_in[7];
    const float* w_in[8] = { (const float*)d_in[8],  (const float*)d_in[9],
                             (const float*)d_in[10], (const float*)d_in[11],
                             (const float*)d_in[12], (const float*)d_in[13],
                             (const float*)d_in[14], (const float*)d_in[15] };
    const float* mw1 = (const float*)d_in[16];
    const float* mb1 = (const float*)d_in[17];
    const float* mw2 = (const float*)d_in[18];
    const float* mb2 = (const float*)d_in[19];
    const unsigned char* self_mask  = (const unsigned char*)d_in[20];
    const unsigned char* cross_mask = (const unsigned char*)d_in[21];

    const size_t NT = (size_t)B_ * ST_ * D_;   // 2,097,152
    size_t cur = 0;
    auto alloc16 = [&](size_t n) { _Float16* p = (_Float16*)d_ws + cur; cur += n; return p; };

    _Float16* wt[8];
    for (int i = 0; i < 8; ++i) wt[i] = alloc16((size_t)D_ * D_);
    _Float16* mw1t  = alloc16((size_t)D_ * MLP_);   // [2048][512]
    _Float16* mw2t  = alloc16((size_t)MLP_ * D_);   // [512][2048]
    _Float16* enc16 = alloc16(NT);
    _Float16* T  = alloc16(NT);
    _Float16* Q  = alloc16(NT);
    _Float16* Kb = alloc16(NT);
    _Float16* V  = alloc16(NT);
    _Float16* O  = alloc16(NT);
    _Float16* Hh = alloc16(NT * 4);                 // [4096][2048]
    float* X = (float*)((_Float16*)d_ws + cur);
    float* Y = X + NT;
    float* OUT = (float*)d_out;

    const int rows = B_ * ST_;                      // 4096
    const dim3 lnGrid(rows / 8), lnBlk(256);
    const dim3 gBlk(128);
    const dim3 gD(rows / 64, D_ / 64);
    const dim3 gM1(rows / 64, MLP_ / 64);
    const dim3 aGrid(ST_ / 64, H_, B_), aBlk(128);
    const float qscale = 0.125f;                    // 1/sqrt(64)

    // ---- one-time weight transpose/convert + encoded convert ----
    for (int i = 0; i < 8; ++i)
        trconv_kernel<<<(D_ * D_) / 256, 256, 0, stream>>>(w_in[i], wt[i], D_, D_);
    trconv_kernel<<<(D_ * MLP_) / 256, 256, 0, stream>>>(mw1, mw1t, D_, MLP_);
    trconv_kernel<<<(MLP_ * D_) / 256, 256, 0, stream>>>(mw2, mw2t, MLP_, D_);
    conv16_kernel<<<NT / 256, 256, 0, stream>>>(encoded, enc16);

    // ---- self attention block ----
    ln512_kernel<<<lnGrid, lnBlk, 0, stream>>>(targets, ln1s, ln1b, T);
    gemm64_kernel<0, _Float16><<<gD, gBlk, 0, stream>>>(T, wt[0], Q,  nullptr, nullptr, D_, D_, qscale);
    gemm64_kernel<0, _Float16><<<gD, gBlk, 0, stream>>>(T, wt[1], Kb, nullptr, nullptr, D_, D_, 1.0f);
    gemm64_kernel<0, _Float16><<<gD, gBlk, 0, stream>>>(T, wt[2], V,  nullptr, nullptr, D_, D_, 1.0f);
    attn64_kernel<<<aGrid, aBlk, 0, stream>>>(Q, Kb, V, O, self_mask, SK_);
    gemm64_kernel<1, float><<<gD, gBlk, 0, stream>>>(O, wt[3], X, nullptr, targets, D_, D_, 1.0f);

    // ---- cross attention block ----
    ln512_kernel<<<lnGrid, lnBlk, 0, stream>>>(X, ln2s, ln2b, T);
    gemm64_kernel<0, _Float16><<<gD, gBlk, 0, stream>>>(T,     wt[4], Q,  nullptr, nullptr, D_, D_, qscale);
    gemm64_kernel<0, _Float16><<<gD, gBlk, 0, stream>>>(enc16, wt[5], Kb, nullptr, nullptr, D_, D_, 1.0f);
    gemm64_kernel<0, _Float16><<<gD, gBlk, 0, stream>>>(enc16, wt[6], V,  nullptr, nullptr, D_, D_, 1.0f);
    attn64_kernel<<<aGrid, aBlk, 0, stream>>>(Q, Kb, V, O, cross_mask, SK_);
    gemm64_kernel<1, float><<<gD, gBlk, 0, stream>>>(O, wt[7], Y, nullptr, X, D_, D_, 1.0f);

    // ---- MLP block ----
    ln512_kernel<<<lnGrid, lnBlk, 0, stream>>>(Y, ln3s, ln3b, T);
    gemm64_kernel<2, _Float16><<<gM1, gBlk, 0, stream>>>(T, mw1t, Hh, mb1, nullptr, D_, MLP_, 1.0f);
    gemm64_kernel<3, float><<<gD, gBlk, 0, stream>>>(Hh, mw2t, OUT, mb2, Y, MLP_, D_, 1.0f);

    (void)in_sizes; (void)n_in; (void)out_size; (void)ws_size;
}
